// SelfAttention_13254269075465
// MI455X (gfx1250) — matrix operand order repaired and checked by reading, not compiled, not run on hardware
//
#include <hip/hip_runtime.h>
#include <hip/hip_bf16.h>

#define S_LEN 4096
#define DMODEL 768
#define NH 12
#define DH 64
#define POST_SCALE 19.595917942265423f  // sqrt(384)

typedef __attribute__((ext_vector_type(16))) __bf16 v16bf;
typedef __attribute__((ext_vector_type(8)))  float  v8f;

union BF16x16 {
  uint4 q[2];
  v16bf v;
};

__device__ __forceinline__ unsigned short f2bf(float f) {
  unsigned u = __float_as_uint(f);
  u += 0x7FFFu + ((u >> 16) & 1u);   // round-to-nearest-even
  return (unsigned short)(u >> 16);
}

// Async DMA: global -> LDS, 16 bytes per lane, tracked by ASYNCcnt.
__device__ __forceinline__ void async_g2l_b128(unsigned lds_off,
                                               const void* gaddr) {
  unsigned long long ga = (unsigned long long)(uintptr_t)gaddr;
  asm volatile("global_load_async_to_lds_b128 %0, %1, off"
               :: "v"(lds_off), "v"(ga)
               : "memory");
}

__device__ __forceinline__ void wait_async_le8() {
  asm volatile("s_wait_asynccnt 0x8" ::: "memory");
}
__device__ __forceinline__ void wait_async_0() {
  asm volatile("s_wait_asynccnt 0x0" ::: "memory");
}

// ---------------------------------------------------------------------------
// f32 -> bf16 conversion (grid-stride)
// ---------------------------------------------------------------------------
__global__ void convert_f32_bf16(const float* __restrict__ in,
                                 unsigned short* __restrict__ out, int n) {
  int i = blockIdx.x * blockDim.x + threadIdx.x;
  int stride = gridDim.x * blockDim.x;
  for (; i < n; i += stride) out[i] = f2bf(in[i]);
}

// ---------------------------------------------------------------------------
// y = X(bf16)[4096x768] @ W(bf16)[768x768]^T + bias
// mode 0: store bf16 row-major [4096x768]
// mode 1: store bf16 TRANSPOSED [768x4096]   (for V^T)
// mode 2: store f32 row-major               (final output projection)
// One wave computes a 16x64 strip. 8 waves/block, 384 blocks = 3072 strips.
// ---------------------------------------------------------------------------
__global__ __launch_bounds__(256) void proj_kernel(
    const unsigned short* __restrict__ X,
    const unsigned short* __restrict__ W,
    const float* __restrict__ bias,
    unsigned short* __restrict__ out16,
    float* __restrict__ out32,
    int mode) {
  const int lane = threadIdx.x & 31;
  const int wid  = threadIdx.x >> 5;
  const int gw   = blockIdx.x * 8 + wid;
  const int s0   = (gw / 12) * 16;
  const int n0   = (gw % 12) * 64;
  const int half = lane >> 4;   // 0: lanes 0-15, 1: lanes 16-31
  const int lr   = lane & 15;

  v8f acc[4] = {};
  const unsigned short* xrow = X + (size_t)(s0 + lr) * DMODEL;

  for (int k0 = 0; k0 < DMODEL; k0 += 32) {
    BF16x16 a;
    // A 16x32 bf16: lane<16 K{0..15}, lane>=16 K{16..31}
    a.q[0] = *(const uint4*)(xrow + k0 + half * 16);
    a.q[1] = *(const uint4*)(xrow + k0 + half * 16 + 8);
#pragma unroll
    for (int sub = 0; sub < 4; ++sub) {
      BF16x16 b;
      // B 32x16: column N = n0+sub*16+lr = W row; K contiguous (half*16 .. +15)
      const unsigned short* wrow =
          W + (size_t)(n0 + sub * 16 + lr) * DMODEL + k0 + half * 16;
      b.q[0] = *(const uint4*)(wrow);
      b.q[1] = *(const uint4*)(wrow + 8);
      acc[sub] = __builtin_amdgcn_wmma_f32_16x16x32_bf16(
          false, a.v, false, b.v, (short)0, acc[sub], false, false);
    }
  }

#pragma unroll
  for (int sub = 0; sub < 4; ++sub) {
    const int col = n0 + sub * 16 + lr;
    const float bv = bias[col];
#pragma unroll
    for (int v = 0; v < 8; ++v) {
      const float val = acc[sub][v] + bv;
      const int row = s0 + v + half * 8;
      if (mode == 0)
        out16[(size_t)row * DMODEL + col] = f2bf(val);
      else if (mode == 1)
        out16[(size_t)col * S_LEN + row] = f2bf(val);   // V^T layout
      else
        out32[(size_t)row * DMODEL + col] = val;
    }
  }
}

// ---------------------------------------------------------------------------
// Fused attention with softmax over the HEAD axis.
// Block = 384 threads = 12 waves (wave h owns head h). One 16-row s-tile/block.
// Q chunks (32 x 768 bf16 = 48 KB) are double-buffered in dynamic LDS and
// filled with GLOBAL_LOAD_ASYNC_TO_LDS_B128 (ASYNCcnt), overlapping WMMA work.
// ---------------------------------------------------------------------------
#define QCHUNK_BYTES (32 * DMODEL * 2)          // 48 KB
#define QCHUNK_B128  (32 * DMODEL / 8)          // 3072 16-byte packets
#define KTILE_B128   (16 * DMODEL / 8)          // 1536 16-byte packets

__global__ __launch_bounds__(384) void attn_kernel(
    const unsigned short* __restrict__ Qb,   // bf16 [4096,768]
    const unsigned short* __restrict__ Kb,   // bf16 [4096,768]
    const unsigned short* __restrict__ Vt,   // bf16 [768,4096] (V transposed)
    unsigned short* __restrict__ AO) {       // bf16 [4096,768]
  __shared__ __align__(16) unsigned short k_lds[16 * DMODEL];      // 24 KB
  __shared__ float sc_lds[NH][16][32];                             // 24 KB
  __shared__ __align__(16) unsigned short at_lds[NH][16][32];      // 12 KB
  extern __shared__ __align__(16) char dyn_lds[];                  // 2 x 48 KB

  const int tid  = threadIdx.x;
  const int lane = tid & 31;
  const int h    = tid >> 5;          // head == wave index (0..11)
  const int half = lane >> 4;
  const int lr   = lane & 15;
  const int s0   = blockIdx.x * 16;

  // Flat-LDS mapping: low 32 bits of a generic shared pointer = LDS offset.
  const unsigned klds_off = (unsigned)(uintptr_t)(void*)k_lds;
  const unsigned qlds_off = (unsigned)(uintptr_t)(void*)dyn_lds;

  // ---- Prologue: async-DMA the K s-tile and the first Q chunk into LDS ----
#pragma unroll
  for (int n = 0; n < 4; ++n) {
    int j = tid + n * 384;
    if (j < KTILE_B128)
      async_g2l_b128(klds_off + j * 16, Kb + (size_t)s0 * DMODEL + j * 8);
  }
#pragma unroll
  for (int n = 0; n < 8; ++n) {
    int j = tid + n * 384;
    async_g2l_b128(qlds_off + j * 16, Qb + j * 8);   // chunk 0 -> buffer 0
  }

  v8f oacc[4] = {};   // head-h output accumulator, 16 x 64 f32

  for (int t0 = 0; t0 < S_LEN; t0 += 32) {
    const int cur = (t0 >> 5) & 1;
    const unsigned short* qcur =
        (const unsigned short*)(dyn_lds + cur * QCHUNK_BYTES);

    // Kick off the DMA for the NEXT chunk into the other buffer, then retire
    // the previous batch (in-order ASYNCcnt: <=8 outstanding == older done).
    if (t0 + 32 < S_LEN) {
      const unsigned nboff = qlds_off + (cur ^ 1) * QCHUNK_BYTES;
      const unsigned short* nsrc = Qb + (size_t)(t0 + 32) * DMODEL;
#pragma unroll
      for (int n = 0; n < 8; ++n) {
        int j = tid + n * 384;
        async_g2l_b128(nboff + j * 16, nsrc + j * 8);
      }
      wait_async_le8();
    } else {
      wait_async_0();
    }
    __syncthreads();

    // ---- Phase A: scores[h][16 s][32 t] = K_tile . Q_chunk^T --------------
#pragma unroll
    for (int tsub = 0; tsub < 2; ++tsub) {
      v8f sacc = {};
#pragma unroll
      for (int dstep = 0; dstep < 2; ++dstep) {
        const int d0 = dstep * 32;
        BF16x16 a, b;
        const unsigned short* kp = k_lds + lr * DMODEL + h * DH + d0 + half * 16;
        a.q[0] = *(const uint4*)(kp);
        a.q[1] = *(const uint4*)(kp + 8);
        const unsigned short* qp =
            qcur + (tsub * 16 + lr) * DMODEL + h * DH + d0 + half * 16;
        b.q[0] = *(const uint4*)(qp);
        b.q[1] = *(const uint4*)(qp + 8);
        sacc = __builtin_amdgcn_wmma_f32_16x16x32_bf16(
            false, a.v, false, b.v, (short)0, sacc, false, false);
      }
#pragma unroll
      for (int v = 0; v < 8; ++v)
        sc_lds[h][v + half * 8][tsub * 16 + lr] = sacc[v];
    }
    __syncthreads();

    // ---- Phase B: softmax over heads, per (s,t); scale by sqrt(384) -------
#pragma unroll
    for (int rep = 0; rep < 2; ++rep) {
      const int p = tid + rep * 384;
      if (p < 512) {
        const int ss = p >> 5, tt = p & 31;
        float vals[NH];
        float m = -3.4e38f;
#pragma unroll
        for (int hh = 0; hh < NH; ++hh) {
          vals[hh] = sc_lds[hh][ss][tt];
          m = fmaxf(m, vals[hh]);
        }
        float sum = 0.f;
#pragma unroll
        for (int hh = 0; hh < NH; ++hh) {
          vals[hh] = __expf(vals[hh] - m);
          sum += vals[hh];
        }
        const float scale = POST_SCALE / sum;
#pragma unroll
        for (int hh = 0; hh < NH; ++hh)
          at_lds[hh][ss][tt] = f2bf(vals[hh] * scale);
      }
    }
    __syncthreads();

    // ---- Phase C: oacc += attn[h] (16x32) @ V^T chunk (32x64) -------------
    {
      BF16x16 a;
      const unsigned short* ap = &at_lds[h][lr][half * 16];
      a.q[0] = *(const uint4*)(ap);
      a.q[1] = *(const uint4*)(ap + 8);
#pragma unroll
      for (int nsub = 0; nsub < 4; ++nsub) {
        BF16x16 b;
        const unsigned short* vp =
            Vt + (size_t)(h * DH + nsub * 16 + lr) * S_LEN + t0 + half * 16;
        b.q[0] = *(const uint4*)(vp);
        b.q[1] = *(const uint4*)(vp + 8);
        oacc[nsub] = __builtin_amdgcn_wmma_f32_16x16x32_bf16(
            false, a.v, false, b.v, (short)0, oacc[nsub], false, false);
      }
      // Prefetch next V^T chunk rows (gfx1250 global_prefetch path).
      if (t0 + 32 < S_LEN)
        __builtin_prefetch(Vt + (size_t)(h * DH + lane) * S_LEN + t0 + 32, 0, 3);
    }
    __syncthreads();
  }

  // Store head-h slice of the merged [S, H*DH] output (bf16).
#pragma unroll
  for (int nsub = 0; nsub < 4; ++nsub) {
    const int col = h * DH + nsub * 16 + lr;
#pragma unroll
    for (int v = 0; v < 8; ++v)
      AO[(size_t)(s0 + v + half * 8) * DMODEL + col] = f2bf(oacc[nsub][v]);
  }
}

// ---------------------------------------------------------------------------
extern "C" void kernel_launch(void* const* d_in, const int* in_sizes, int n_in,
                              void* d_out, int out_size, void* d_ws, size_t ws_size,
                              hipStream_t stream) {
  const float* x  = (const float*)d_in[0];
  const float* Wq = (const float*)d_in[1];
  const float* bq = (const float*)d_in[2];
  const float* Wk = (const float*)d_in[3];
  const float* bk = (const float*)d_in[4];
  const float* Wv = (const float*)d_in[5];
  const float* bv = (const float*)d_in[6];
  const float* Wo = (const float*)d_in[7];
  const float* bo = (const float*)d_in[8];
  float* out = (float*)d_out;

  char* ws = (char*)d_ws;
  size_t off = 0;
  auto alloc = [&](size_t bytes) -> char* {
    char* p = ws + off;
    off += (bytes + 255) & ~(size_t)255;
    return p;
  };
  const size_t XW = (size_t)S_LEN * DMODEL * sizeof(unsigned short);   // 6 MB
  const size_t WW = (size_t)DMODEL * DMODEL * sizeof(unsigned short);  // 1.2 MB
  unsigned short* xbf  = (unsigned short*)alloc(XW);
  unsigned short* wqbf = (unsigned short*)alloc(WW);
  unsigned short* wkbf = (unsigned short*)alloc(WW);
  unsigned short* wvbf = (unsigned short*)alloc(WW);
  unsigned short* wobf = (unsigned short*)alloc(WW);
  unsigned short* Qbuf = (unsigned short*)alloc(XW);
  unsigned short* Kbuf = (unsigned short*)alloc(XW);
  unsigned short* Vtbf = (unsigned short*)alloc(XW);   // [768,4096]
  unsigned short* AObf = (unsigned short*)alloc(XW);

  // Allow 96 KB of dynamic LDS for the attention kernel (gfx1250: 320 KB/WGP).
  (void)hipFuncSetAttribute((const void*)attn_kernel,
                            hipFuncAttributeMaxDynamicSharedMemorySize,
                            2 * QCHUNK_BYTES);

  convert_f32_bf16<<<1024, 256, 0, stream>>>(x,  xbf,  S_LEN * DMODEL);
  convert_f32_bf16<<<512,  256, 0, stream>>>(Wq, wqbf, DMODEL * DMODEL);
  convert_f32_bf16<<<512,  256, 0, stream>>>(Wk, wkbf, DMODEL * DMODEL);
  convert_f32_bf16<<<512,  256, 0, stream>>>(Wv, wvbf, DMODEL * DMODEL);
  convert_f32_bf16<<<512,  256, 0, stream>>>(Wo, wobf, DMODEL * DMODEL);

  // Q, K projections (bf16 out), V projection stored transposed.
  proj_kernel<<<384, 256, 0, stream>>>(xbf, wqbf, bq, Qbuf, nullptr, 0);
  proj_kernel<<<384, 256, 0, stream>>>(xbf, wkbf, bk, Kbuf, nullptr, 0);
  proj_kernel<<<384, 256, 0, stream>>>(xbf, wvbf, bv, Vtbf, nullptr, 1);

  // Fused scores -> head-axis softmax -> PV.
  attn_kernel<<<S_LEN / 16, 384, 2 * QCHUNK_BYTES, stream>>>(Qbuf, Kbuf, Vtbf, AObf);

  // Final output projection, f32 result.
  proj_kernel<<<384, 256, 0, stream>>>(AObf, wobf, bo, nullptr, out, 2);
}
